// Distillation_36481452212467
// MI455X (gfx1250) — matrix operand order repaired and checked by reading, not compiled, not run on hardware
//
#include <hip/hip_runtime.h>

#define WAY   100
#define SHOT  5
#define DIM   2048
#define NQ    10000
#define WP    112      // ways padded to 7*16 for WMMA tiling
#define KVAL  512
#define TK    32       // bf16 WMMA K-step
#define KC    128      // K-chunk staged in LDS per barrier (4 WMMA k-steps)
#define BM    32       // rows per block (two 16-row WMMA sub-tiles)

typedef __bf16 bf16_t;
typedef __attribute__((ext_vector_type(16))) __bf16 v16bf;
typedef __attribute__((ext_vector_type(8)))  float  v8f;

struct TopK { unsigned thrKey; int ties; int tiectr; float kf; };

// ---------------- helpers ----------------

__device__ inline bf16_t f2bf(float f) {
    union { float f; unsigned u; } x; x.f = f;
    unsigned r = x.u + 0x7FFFu + ((x.u >> 16) & 1u);
    unsigned short h = (unsigned short)(r >> 16);
    bf16_t out;
    __builtin_memcpy(&out, &h, 2);
    return out;
}

// round-to-nearest-even pack of two floats into two bf16 (lo, hi)
__device__ inline unsigned pkbf2(float a, float b) {
    union { float f; unsigned u; } xa, xb; xa.f = a; xb.f = b;
    unsigned ra = xa.u + 0x7FFFu + ((xa.u >> 16) & 1u);
    unsigned rb = xb.u + 0x7FFFu + ((xb.u >> 16) & 1u);
    return (ra >> 16) | (rb & 0xFFFF0000u);
}

__device__ inline unsigned ordkey(float v) {
    unsigned u = __float_as_uint(v);
    return (u & 0x80000000u) ? ~u : (u | 0x80000000u);
}

// ---------------- setup kernels ----------------

__global__ void set_int(int* p, int v) { *p = v; }

__global__ __launch_bounds__(256) void iota_idx(int* __restrict__ p) {
    int i = blockIdx.x * 256 + threadIdx.x;
    if (i < NQ) p[i] = i;
}

// one block per (padded) way: mean over shots, L2-normalize, emit fp32 + bf16
__global__ __launch_bounds__(256) void proto_from_shots(
    const float* __restrict__ shots, float* __restrict__ protoF, bf16_t* __restrict__ protoB) {
    int w = blockIdx.x, t = threadIdx.x;
    __shared__ float red[256];
    if (w >= WAY) {
        for (int d = t; d < DIM; d += 256) {
            protoF[(size_t)w * DIM + d] = 0.f;
            protoB[(size_t)w * DIM + d] = f2bf(0.f);
        }
        return;
    }
    const float* base = shots + (size_t)w * SHOT * DIM;
    float ss = 0.f;
    for (int d = t; d < DIM; d += 256) {
        float m = 0.f;
        for (int s = 0; s < SHOT; ++s) m += base[(size_t)s * DIM + d];
        m *= (1.f / SHOT);
        protoF[(size_t)w * DIM + d] = m;
        ss += m * m;
    }
    red[t] = ss; __syncthreads();
    for (int s = 128; s > 0; s >>= 1) { if (t < s) red[t] += red[t + s]; __syncthreads(); }
    float inv = 1.f / fmaxf(sqrtf(red[0]), 1e-12f);
    for (int d = t; d < DIM; d += 256) {
        float v = protoF[(size_t)w * DIM + d] * inv;
        protoF[(size_t)w * DIM + d] = v;
        protoB[(size_t)w * DIM + d] = f2bf(v);
    }
}

// one block per query row: 1/max(||q||,1e-12)
__global__ __launch_bounds__(256) void row_invnorm(const float* __restrict__ q, float* __restrict__ invn) {
    int r = blockIdx.x, t = threadIdx.x;
    __shared__ float red[256];
    const float* row = q + (size_t)r * DIM;
    float ss = 0.f;
    for (int d = t; d < DIM; d += 256) { float v = row[d]; ss += v * v; }
    red[t] = ss; __syncthreads();
    for (int s = 128; s > 0; s >>= 1) { if (t < s) red[t] += red[t + s]; __syncthreads(); }
    if (t == 0) invn[r] = 1.f / fmaxf(sqrtf(red[0]), 1e-12f);
}

// ---------------- WMMA GEMM: logits[q][w] = <feat_n[q], proto[w]> ----------------
// Block: 32 rows x 112 cols, 7 waves; wave i owns the two 16x16 tiles in column 16*i.
// Row indirection (index, row pointer, inv-norm) hoisted out of the K loop; ping-pong
// LDS staging overlaps chunk c+1 loads with the 8 WMMAs of chunk c; one barrier/chunk.
// B fragments stream straight from L2-resident bf16 proto (contiguous 32B per lane).
__global__ __launch_bounds__(224) void gemm_wmma(
    const float* __restrict__ qdata, const float* __restrict__ invn,
    const int* __restrict__ idxArr, const int* __restrict__ pN,
    const bf16_t* __restrict__ protoB, float* __restrict__ out) {
    const int n  = *pN;
    const int q0 = blockIdx.x * BM;
    if (q0 >= n) return;
    const int tid  = threadIdx.x;
    const int wave = tid >> 5;
    const int lane = tid & 31;
    const int w0   = wave * 16;

    __shared__ bf16_t As[2][BM][KC];   // 16 KB ping-pong

    v8f acc0 = {};                  // rows q0 .. q0+15
    v8f acc1 = {};                  // rows q0+16 .. q0+31
    const int m   = lane & 15;
    const int hi  = lane >> 4;
    const int kbA = hi ? 16 : 0;    // A-fragment k-base within 32
    const int nB  = w0 + m;
    const int kbB = hi ? 16 : 0;    // B-fragment k-base within 32

    constexpr int NU  = BM * KC / 4;          // 1024 float4 units per chunk
    constexpr int UN  = (NU + 223) / 224;     // 5 units per thread
    constexpr int NCH = DIM / KC;             // 16 chunks

    // K-invariant per-unit state: row base pointer + inverse norm (0 => zero-fill)
    const float* rowPtr[UN];
    float        sval[UN];
    #pragma unroll
    for (int it = 0; it < UN; ++it) {
        int u = tid + it * 224;
        rowPtr[it] = qdata;
        sval[it]   = 0.f;
        if (u < NU) {
            int qq = q0 + (u >> 5);           // KC/4 = 32 float4 per row
            if (qq < n) {
                int row = idxArr[qq];
                rowPtr[it] = qdata + (size_t)row * DIM;
                sval[it]   = invn[row];
            }
        }
    }

    auto stage = [&](int buf, int kt) {
        float4 v[UN];
        #pragma unroll
        for (int it = 0; it < UN; ++it) {
            int u = tid + it * 224;
            v[it].x = v[it].y = v[it].z = v[it].w = 0.f;
            if (u < NU) {
                int c4 = (u & 31) << 2;
                v[it] = *reinterpret_cast<const float4*>(rowPtr[it] + kt + c4);
                if (it == 0 && kt + KC < DIM)
                    __builtin_prefetch(rowPtr[it] + kt + KC, 0, 0);  // global_prefetch_b8
            }
        }
        #pragma unroll
        for (int it = 0; it < UN; ++it) {
            int u = tid + it * 224;
            if (u < NU) {
                int r  = u >> 5;
                int c4 = (u & 31) << 2;
                uint2 p;
                p.x = pkbf2(v[it].x * sval[it], v[it].y * sval[it]);
                p.y = pkbf2(v[it].z * sval[it], v[it].w * sval[it]);
                *reinterpret_cast<uint2*>(&As[buf][r][c4]) = p;
            }
        }
    };

    stage(0, 0);
    __syncthreads();
    for (int c = 0; c < NCH; ++c) {
        const int cur = c & 1;
        if (c + 1 < NCH) stage(cur ^ 1, (c + 1) * KC);
        const int kt = c * KC;
        #pragma unroll
        for (int kk = 0; kk < KC; kk += TK) {
            v16bf b;
            const bf16_t* bp = protoB + (size_t)nB * DIM + kt + kk + kbB;
            #pragma unroll
            for (int e = 0; e < 16; ++e) b[e] = bp[e];

            v16bf a0, a1;
            #pragma unroll
            for (int i = 0; i < 16; ++i) {
                int g = i >> 3, j = i & 7;
                int kcol = kk + kbA + i;
                a0[i] = As[cur][m][kcol];
                a1[i] = As[cur][16 + m][kcol];
            }
            acc0 = __builtin_amdgcn_wmma_f32_16x16x32_bf16(
                false, a0, false, b, (short)0, acc0, false, false);
            acc1 = __builtin_amdgcn_wmma_f32_16x16x32_bf16(
                false, a1, false, b, (short)0, acc1, false, false);
        }
        __syncthreads();
    }
    // C/D layout: VGPR r -> M = r + 8*hi, N = lane&15
    #pragma unroll
    for (int r = 0; r < 8; ++r) {
        int mm = r + (hi ? 8 : 0);
        int qq = q0 + mm;
        if (qq < n) out[(size_t)qq * WP + w0 + m] = acc0[r];
        qq = q0 + 16 + mm;
        if (qq < n) out[(size_t)qq * WP + w0 + m] = acc1[r];
    }
}

// ---------------- logits post-processing ----------------

__global__ __launch_bounds__(256) void col_sumsq(const float* __restrict__ L, const int* __restrict__ pN,
                                                 float* __restrict__ colss) {
    int w = blockIdx.x, t = threadIdx.x, n = *pN;
    __shared__ float red[256];
    float ss = 0.f;
    for (int i = t; i < n; i += 256) { float v = L[(size_t)i * WP + w]; ss += v * v; }
    red[t] = ss; __syncthreads();
    for (int s = 128; s > 0; s >>= 1) { if (t < s) red[t] += red[t + s]; __syncthreads(); }
    if (t == 0) colss[w] = red[0];
}

__global__ __launch_bounds__(256) void scale_rowstats(float* __restrict__ L, const int* __restrict__ pN,
                                                      const float* __restrict__ colss,
                                                      const float* __restrict__ tpp,
                                                      float* __restrict__ acert, int* __restrict__ amax) {
    __shared__ float cn[WP];
    int t = threadIdx.x;
    for (int w = t; w < WP; w += 256) cn[w] = sqrtf(colss[w]) + 1e-6f;
    __syncthreads();
    int n = *pN, q = blockIdx.x * 256 + t;
    if (q >= n) return;
    float tp = *tpp;
    float best = -3.4e38f; int bi = 0;
    float* row = L + (size_t)q * WP;
    for (int w = 0; w < WAY; ++w) {
        float v = row[w] / cn[w] * tp;
        row[w] = v;
        if (v > best) { best = v; bi = w; }
    }
    acert[q] = best; amax[q] = bi;
}

// exact k-th largest via 32-pass MSB radix select on ordered float keys
__global__ __launch_bounds__(1024) void topk_select(const float* __restrict__ vals,
                                                    const int* __restrict__ pN, TopK* tk) {
    int n = *pN, t = threadIdx.x;
    __shared__ int sr[1024];
    __shared__ unsigned s_pv;
    __shared__ int s_rem;
    int k = n < KVAL ? n : KVAL;
    if (t == 0) { s_pv = 0u; s_rem = k; }
    __syncthreads();
    if (n > 0) {
        for (int bit = 31; bit >= 0; --bit) {
            unsigned pv = s_pv; int rem = s_rem;
            unsigned want = (pv >> bit) | 1u;
            int c = 0;
            for (int i = t; i < n; i += 1024)
                if ((ordkey(vals[i]) >> bit) == want) c++;
            sr[t] = c; __syncthreads();
            for (int s = 512; s > 0; s >>= 1) { if (t < s) sr[t] += sr[t + s]; __syncthreads(); }
            if (t == 0) { if (sr[0] >= rem) s_pv = pv | (1u << bit); else s_rem = rem - sr[0]; }
            __syncthreads();
        }
    }
    unsigned pv = s_pv;
    int cg = 0;
    for (int i = t; i < n; i += 1024) if (ordkey(vals[i]) > pv) cg++;
    sr[t] = cg; __syncthreads();
    for (int s = 512; s > 0; s >>= 1) { if (t < s) sr[t] += sr[t + s]; __syncthreads(); }
    if (t == 0) {
        tk->thrKey = (n > 0) ? pv : 0xFFFFFFFFu;
        tk->ties   = k - sr[0];
        tk->tiectr = 0;
        tk->kf     = (float)(k > 0 ? k : 1);
    }
}

__global__ __launch_bounds__(256) void mark_selected(const float* __restrict__ vals,
                                                     const int* __restrict__ pN,
                                                     TopK* tk, int* __restrict__ sel) {
    int n = *pN, q = blockIdx.x * 256 + threadIdx.x;
    if (q >= n) return;
    unsigned key = ordkey(vals[q]);
    int s = 0;
    if (key > tk->thrKey) s = 1;
    else if (key == tk->thrKey) { if (atomicAdd(&tk->tiectr, 1) < tk->ties) s = 1; }
    sel[q] = s;
}

// block-per-query atomic scatter: add[pseudo] += abs_cert * feat_n / k
__global__ __launch_bounds__(256) void refine_scatter(
    const float* __restrict__ qdata, const float* __restrict__ invn,
    const int* __restrict__ idxArr, const int* __restrict__ pN,
    const int* __restrict__ sel, const int* __restrict__ amax,
    const float* __restrict__ acert, const TopK* tk, float* __restrict__ add) {
    int n = *pN, q = blockIdx.x;
    if (q >= n || !sel[q]) return;
    int row = idxArr[q];
    int w = amax[q];
    float coef = acert[q] / tk->kf * invn[row];
    const float* src = qdata + (size_t)row * DIM;
    float* dst = add + (size_t)w * DIM;
    for (int d = threadIdx.x; d < DIM; d += 256) atomicAdd(&dst[d], coef * src[d]);
}

__global__ __launch_bounds__(256) void proto_combine(const float* __restrict__ protoF,
                                                     const float* __restrict__ add,
                                                     bf16_t* __restrict__ protoB) {
    int w = blockIdx.x;
    for (int d = threadIdx.x; d < DIM; d += 256)
        protoB[(size_t)w * DIM + d] = f2bf(protoF[(size_t)w * DIM + d] + add[(size_t)w * DIM + d]);
}

// abs_c = rowmax; rel_c = sum p*log(p+1e-10) over ways
__global__ __launch_bounds__(256) void certainty(const float* __restrict__ L, const int* __restrict__ pN,
                                                 float* __restrict__ absc, float* __restrict__ relc) {
    int n = *pN, q = blockIdx.x * 256 + threadIdx.x;
    if (q >= n) return;
    const float* row = L + (size_t)q * WP;
    float m = -3.4e38f;
    for (int w = 0; w < WAY; ++w) m = fmaxf(m, row[w]);
    float se = 0.f;
    for (int w = 0; w < WAY; ++w) se += expf(row[w] - m);
    float rel = 0.f;
    for (int w = 0; w < WAY; ++w) { float p = expf(row[w] - m) / se; rel += p * logf(p + 1e-10f); }
    absc[q] = m; relc[q] = rel;
}

// deterministic ascending compaction of diff>0 / diff<0 (single block scan)
__global__ __launch_bounds__(1024) void compact_diff(
    const float* __restrict__ ac, const float* __restrict__ rc,
    const float* __restrict__ acd, const float* __restrict__ rcd,
    int* __restrict__ resIdx, int* __restrict__ dctIdx, int* __restrict__ cnts) {
    __shared__ int sc[1024];
    __shared__ int base0, base1;
    int t = threadIdx.x;
    if (t == 0) { base0 = 0; base1 = 0; }
    __syncthreads();
    for (int start = 0; start < NQ; start += 1024) {
        int i = start + t;
        bool valid = i < NQ;
        float d = valid ? (ac[i] - acd[i] + rc[i] - rcd[i]) : 0.f;
        // res pass (diff > 0)
        int f = (valid && d > 0.f) ? 1 : 0;
        sc[t] = f; __syncthreads();
        for (int off = 1; off < 1024; off <<= 1) {
            int v = (t >= off) ? sc[t - off] : 0; __syncthreads();
            sc[t] += v; __syncthreads();
        }
        if (f) resIdx[base0 + sc[t] - 1] = i;
        __syncthreads();
        if (t == 0) base0 += sc[1023];
        __syncthreads();
        // dct pass (diff < 0)
        f = (valid && d < 0.f) ? 1 : 0;
        sc[t] = f; __syncthreads();
        for (int off = 1; off < 1024; off <<= 1) {
            int v = (t >= off) ? sc[t - off] : 0; __syncthreads();
            sc[t] += v; __syncthreads();
        }
        if (f) dctIdx[base1 + sc[t] - 1] = i;
        __syncthreads();
        if (t == 0) base1 += sc[1023];
        __syncthreads();
    }
    if (t == 0) { cnts[0] = base0; cnts[1] = base1; }
}

__global__ __launch_bounds__(256) void write_indices(const int* __restrict__ resIdx,
                                                     const int* __restrict__ dctIdx,
                                                     const int* __restrict__ cnts, float* __restrict__ out) {
    int i = blockIdx.x * 256 + threadIdx.x;
    int nr = cnts[0], nd = cnts[1];
    if (i < nr) out[2 + i] = (float)resIdx[i];
    if (i < nd) out[2 + nr + i] = (float)dctIdx[i];
}

__global__ __launch_bounds__(256) void write_weights(const float* __restrict__ ac,
                                                     const float* __restrict__ acd,
                                                     const int* __restrict__ cnts, float* __restrict__ out) {
    int q = blockIdx.x * 256 + threadIdx.x;
    if (q >= NQ) return;
    int base = 2 + cnts[0] + cnts[1];
    float a = ac[q], b = acd[q];
    float den = fmaxf(a + b, 1e-8f);
    out[base + q]       = a / den;
    out[base + NQ + q]  = b / den;
}

// ---------------- distill loss (column softmax over query dim) ----------------

__global__ __launch_bounds__(256) void distill_colstats(
    const float* __restrict__ Ll, const float* __restrict__ Lh, const int* __restrict__ pN,
    float* __restrict__ ml, float* __restrict__ sl, float* __restrict__ mh, float* __restrict__ sh) {
    int w = blockIdx.x, t = threadIdx.x, n = *pN;
    __shared__ float r1[256], r2[256];
    float m1 = -3.4e38f, m2 = -3.4e38f;
    for (int i = t; i < n; i += 256) {
        m1 = fmaxf(m1, Ll[(size_t)i * WP + w]);
        m2 = fmaxf(m2, Lh[(size_t)i * WP + w]);
    }
    r1[t] = m1; r2[t] = m2; __syncthreads();
    for (int s = 128; s > 0; s >>= 1) {
        if (t < s) { r1[t] = fmaxf(r1[t], r1[t + s]); r2[t] = fmaxf(r2[t], r2[t + s]); }
        __syncthreads();
    }
    m1 = r1[0]; m2 = r2[0]; __syncthreads();
    float a = 0.f, b = 0.f;
    for (int i = t; i < n; i += 256) {
        a += expf(Ll[(size_t)i * WP + w] - m1);
        b += expf(Lh[(size_t)i * WP + w] - m2);
    }
    r1[t] = a; r2[t] = b; __syncthreads();
    for (int s = 128; s > 0; s >>= 1) {
        if (t < s) { r1[t] += r1[t + s]; r2[t] += r2[t + s]; }
        __syncthreads();
    }
    if (t == 0) { ml[w] = m1; sl[w] = r1[0]; mh[w] = m2; sh[w] = r2[0]; }
}

__global__ __launch_bounds__(256) void distill_accum(
    const float* __restrict__ Ll, const float* __restrict__ Lh, const int* __restrict__ pN,
    const int* __restrict__ idxArr, const float* __restrict__ wsrc,
    const float* __restrict__ ml, const float* __restrict__ sl,
    const float* __restrict__ mh, const float* __restrict__ sh, float* __restrict__ accb) {
    __shared__ float sml[WAY], ssl[WAY], smh[WAY], ssh[WAY];
    int t = threadIdx.x;
    for (int w = t; w < WAY; w += 256) {
        sml[w] = ml[w]; ssl[w] = logf(sl[w]);
        smh[w] = mh[w]; ssh[w] = logf(sh[w]);
    }
    __syncthreads();
    int n = *pN, i = blockIdx.x * 256 + t;
    if (i >= n) return;
    float wgt = wsrc[idxArr[i]];
    float part = 0.f;
    for (int w = 0; w < WAY; ++w) {
        float lpl = Ll[(size_t)i * WP + w] - sml[w] - ssl[w];
        float lph = Lh[(size_t)i * WP + w] - smh[w] - ssh[w];
        float p = expf(lph);
        part += p * (lph - lpl);
    }
    atomicAdd(&accb[0], part * wgt);
    atomicAdd(&accb[1], wgt);
}

__global__ void distill_finalize(const float* accb, const float* tempPtr, float* outSlot) {
    *outSlot = (*tempPtr) * (accb[0] / (accb[1] + 1e-8f));
}

// ---------------- host launcher ----------------

extern "C" void kernel_launch(void* const* d_in, const int* in_sizes, int n_in,
                              void* d_out, int out_size, void* d_ws, size_t ws_size,
                              hipStream_t stream) {
    (void)in_sizes; (void)n_in; (void)out_size; (void)ws_size;
    const float* x_shot    = (const float*)d_in[0];
    const float* x_query   = (const float*)d_in[1];
    const float* dct_shot  = (const float*)d_in[2];
    const float* dct_query = (const float*)d_in[3];
    const float* t_proto   = (const float*)d_in[4];
    const float* t_res     = (const float*)d_in[5];
    const float* t_dct     = (const float*)d_in[6];
    float* out = (float*)d_out;

    char* ws = (char*)d_ws;
    size_t off = 0;
    auto carve = [&](size_t bytes) -> char* {
        char* p = ws + off;
        off = (off + bytes + 255) & ~(size_t)255;
        return p;
    };

    float*  protoRF  = (float*)carve((size_t)WP * DIM * 4);
    float*  protoDF  = (float*)carve((size_t)WP * DIM * 4);
    bf16_t* protoRB  = (bf16_t*)carve((size_t)WP * DIM * 2);
    bf16_t* protoDB  = (bf16_t*)carve((size_t)WP * DIM * 2);
    bf16_t* protoXB  = (bf16_t*)carve((size_t)WP * DIM * 2);  // refined proto scratch
    float*  addBuf   = (float*)carve((size_t)WP * DIM * 4);
    float*  invnR    = (float*)carve((size_t)NQ * 4);
    float*  invnD    = (float*)carve((size_t)NQ * 4);
    float*  LW       = (float*)carve((size_t)10016 * WP * 4);
    float*  LA       = (float*)carve((size_t)10016 * WP * 4);
    float*  LB       = (float*)carve((size_t)10016 * WP * 4);
    float*  colss    = (float*)carve((size_t)WP * 4);
    float*  acert    = (float*)carve((size_t)10016 * 4);
    int*    amax     = (int*)carve((size_t)10016 * 4);
    int*    selfl    = (int*)carve((size_t)10016 * 4);
    float*  absC     = (float*)carve((size_t)NQ * 4);
    float*  relC     = (float*)carve((size_t)NQ * 4);
    float*  absCd    = (float*)carve((size_t)NQ * 4);
    float*  relCd    = (float*)carve((size_t)NQ * 4);
    int*    resIdx   = (int*)carve((size_t)NQ * 4);
    int*    dctIdx   = (int*)carve((size_t)NQ * 4);
    int*    idxIdent = (int*)carve((size_t)NQ * 4);
    int*    cnts     = (int*)carve(8);
    int*    nFull    = (int*)carve(4);
    TopK*   tk       = (TopK*)carve(sizeof(TopK));
    float*  mlA      = (float*)carve((size_t)WAY * 4);
    float*  slA      = (float*)carve((size_t)WAY * 4);
    float*  mhA      = (float*)carve((size_t)WAY * 4);
    float*  shA      = (float*)carve((size_t)WAY * 4);
    float*  accb     = (float*)carve(8);

    const int GROW  = (NQ + 255) / 256;          // 40 row-blocks
    const int GTIL  = (NQ + BM - 1) / BM;        // 313 gemm row-tiles

    set_int<<<1, 1, 0, stream>>>(nFull, NQ);
    iota_idx<<<GROW, 256, 0, stream>>>(idxIdent);
    proto_from_shots<<<WP, 256, 0, stream>>>(x_shot,   protoRF, protoRB);
    proto_from_shots<<<WP, 256, 0, stream>>>(dct_shot, protoDF, protoDB);
    row_invnorm<<<NQ, 256, 0, stream>>>(x_query,   invnR);
    row_invnorm<<<NQ, 256, 0, stream>>>(dct_query, invnD);

    auto distance = [&](const float* qdata, const float* invn, const int* idxArr, const int* pN,
                        const float* protoF, const bf16_t* protoB, float* L) {
        gemm_wmma<<<GTIL, 224, 0, stream>>>(qdata, invn, idxArr, pN, protoB, L);
        col_sumsq<<<WP, 256, 0, stream>>>(L, pN, colss);
        scale_rowstats<<<GROW, 256, 0, stream>>>(L, pN, colss, t_proto, acert, amax);
        topk_select<<<1, 1024, 0, stream>>>(acert, pN, tk);
        mark_selected<<<GROW, 256, 0, stream>>>(acert, pN, tk, selfl);
        hipMemsetAsync(addBuf, 0, (size_t)WP * DIM * 4, stream);
        refine_scatter<<<NQ, 256, 0, stream>>>(qdata, invn, idxArr, pN, selfl, amax, acert, tk, addBuf);
        proto_combine<<<WP, 256, 0, stream>>>(protoF, addBuf, protoXB);
        gemm_wmma<<<GTIL, 224, 0, stream>>>(qdata, invn, idxArr, pN, protoXB, L);
        col_sumsq<<<WP, 256, 0, stream>>>(L, pN, colss);
        scale_rowstats<<<GROW, 256, 0, stream>>>(L, pN, colss, t_proto, acert, amax);
    };

    auto distill = [&](const float* Llow, const float* Lhigh, const int* idxArr, const int* pN,
                       const float* wsrc, const float* tempPtr, float* outSlot) {
        hipMemsetAsync(accb, 0, 8, stream);
        distill_colstats<<<WAY, 256, 0, stream>>>(Llow, Lhigh, pN, mlA, slA, mhA, shA);
        distill_accum<<<GROW, 256, 0, stream>>>(Llow, Lhigh, pN, idxArr, wsrc, mlA, slA, mhA, shA, accb);
        distill_finalize<<<1, 1, 0, stream>>>(accb, tempPtr, outSlot);
    };

    // full-set certainties
    distance(x_query, invnR, idxIdent, nFull, protoRF, protoRB, LW);
    certainty<<<GROW, 256, 0, stream>>>(LW, nFull, absC, relC);
    distance(dct_query, invnD, idxIdent, nFull, protoDF, protoDB, LW);
    certainty<<<GROW, 256, 0, stream>>>(LW, nFull, absCd, relCd);

    // split queries and emit index / weight outputs
    compact_diff<<<1, 1024, 0, stream>>>(absC, relC, absCd, relCd, resIdx, dctIdx, cnts);
    write_indices<<<GROW, 256, 0, stream>>>(resIdx, dctIdx, cnts, out);
    write_weights<<<GROW, 256, 0, stream>>>(absC, absCd, cnts, out);

    // loss_dctl = temp_dct * KL(ldl || lrh) weighted by abs_c[res_idx]  -> out[1]
    distance(x_query,   invnR, resIdx, &cnts[0], protoRF, protoRB, LA);  // lrh
    distance(dct_query, invnD, resIdx, &cnts[0], protoDF, protoDB, LB);  // ldl
    distill(LB, LA, resIdx, &cnts[0], absC, t_dct, out + 1);

    // loss_resl = temp * KL(lrl || ldh) weighted by abs_cd[dct_idx]     -> out[0]
    distance(dct_query, invnD, dctIdx, &cnts[1], protoDF, protoDB, LA);  // ldh
    distance(x_query,   invnR, dctIdx, &cnts[1], protoRF, protoRB, LB);  // lrl
    distill(LB, LA, dctIdx, &cnts[1], absCd, t_res, out + 0);
}
